// DMP_Model_34342558499354
// MI455X (gfx1250) — compile-verified, matrix operand order checked by reading, not run on hardware
//
#include <hip/hip_runtime.h>
#include <hip/hip_bf16.h>

// ---------------- problem constants (match reference) ----------------
#define NN   20000      // nodes
#define DD   512        // input dim
#define HH   256        // hidden dim
#define CC   7          // classes / branches
#define LL   2          // layers
#define EE   131072     // edges per class
#define BB   8192       // pairs
#define F1   384        // 3H/2
#define F2   192        // 3H/4
#define F3   16         // H/16
#define KC   112        // C * H/16
#define EPS  1e-5f

typedef __bf16 bf16_t;
typedef __attribute__((ext_vector_type(16))) __bf16 v16bf;
typedef __attribute__((ext_vector_type(8)))  float  v8f;
typedef __attribute__((ext_vector_type(4)))  unsigned int u32x4;

// ---------------- helpers ----------------
__device__ __forceinline__ unsigned short f2bf(float x) {
    unsigned int u = __float_as_uint(x);
    unsigned int r = u + 0x7FFFu + ((u >> 16) & 1u);   // round to nearest even
    return (unsigned short)(r >> 16);
}

// Load one 16x32 bf16 fragment (A-layout, ISA 7.12.2):
//   lanes 0-15 : row = r0+lane,    K = k0+{0..7, 16..23}
//   lanes 16-31: row = r0+lane-16, K = k0+{8..15,24..31}
// base is row-major [rows, ld] bf16 (stored as ushort), ld multiple of 8,
// base 16B aligned -> two b128 loads per lane.
__device__ __forceinline__ v16bf load_frag(const unsigned short* base, int ld,
                                           int r0, int k0, int lane) {
    int r  = r0 + (lane & 15);
    int kb = k0 + ((lane >> 4) << 3);
    const unsigned short* p = base + (size_t)r * ld + kb;
    union { v16bf v; u32x4 u[2]; } f;
    f.u[0] = *(const u32x4*)(p);
    f.u[1] = *(const u32x4*)(p + 16);
    return f.v;
}

__device__ __forceinline__ v8f wmma_bf16(v16bf a, v16bf b, v8f c) {
    return __builtin_amdgcn_wmma_f32_16x16x32_bf16(false, a, false, b,
                                                   (short)0, c, false, false);
}

// epilogue for one 16x16 tile (D layout: lane<16 -> (M=r,N=lane);
// lane>=16 -> (M=r+8,N=lane-16))
__device__ __forceinline__ void store_tile(v8f acc, float* Cf, unsigned short* Cb,
                                           const float* bias, int mb, int n,
                                           int ldc, int ccol0, int relu) {
    float bv = bias ? bias[n] : 0.0f;
#pragma unroll
    for (int r = 0; r < 8; ++r) {
        float v = acc[r] + bv;
        if (relu) v = fmaxf(v, 0.0f);
        size_t idx = (size_t)(mb + r) * ldc + ccol0 + n;
        if (Cf) Cf[idx] = v;
        if (Cb) Cb[idx] = f2bf(v);
    }
}

// ---------------- generic bf16 WMMA GEMM ----------------
// C[M,N] = A[M,K] @ B[K,N] (+bias) (relu?) ; Bt given as [N,K] bf16.
// Each wave owns a 16 x (16*NSUB) stripe: NSUB N-tiles share one A fragment
// per K-step (compile-time NSUB -> straight-line unrolled inner loop).
// block = 8 waves = 8 M-tiles. grid.x = N / (16*NSUB)  (exact).
template <int NSUB>
__global__ __launch_bounds__(256)
void k_gemm_bf16(const unsigned short* __restrict__ A,
                 const unsigned short* __restrict__ Bt,
                 float* __restrict__ Cf, unsigned short* __restrict__ Cb,
                 const float* __restrict__ bias,
                 int M, int K, int ldc, int ccol0, int relu) {
    int wave = threadIdx.x >> 5;
    int lane = threadIdx.x & 31;
    int m0 = (blockIdx.y * 8 + wave) * 16;      // wave-uniform guard -> EXEC all 1s
    if (m0 >= M) return;
    int n0 = blockIdx.x * (16 * NSUB);

    v8f acc[NSUB] = {};
    for (int k0 = 0; k0 < K; k0 += 32) {
        v16bf a = load_frag(A, K, m0, k0, lane);
        // prefetch next A k-step (speculative; emits global_prefetch_b8)
        __builtin_prefetch(A + (size_t)(m0 + (lane & 15)) * K + k0 + 32, 0, 1);
#pragma unroll
        for (int t = 0; t < NSUB; ++t) {
            v16bf b = load_frag(Bt, K, n0 + 16 * t, k0, lane);
            acc[t] = wmma_bf16(a, b, acc[t]);
        }
    }
    int nl = lane & 15;
    int mb = m0 + ((lane >> 4) << 3);
#pragma unroll
    for (int t = 0; t < NSUB; ++t)
        store_tile(acc[t], Cf, Cb, bias, mb, n0 + 16 * t + nl, ldc, ccol0, relu);
}

// ---------------- fused comp-GCN edge kernel ----------------
// Per 16-edge tile: A = bf16(t[col]*t[row]) [16,256] in LDS,
// D = A @ dpWt^T (WMMA), then tnext[row[e], :] += D + dp_b  (atomics).
// A fragments are cached in registers (8 x v16bf = 64 VGPRs) and reused
// across the wave's 2 N-tiles.
__global__ __launch_bounds__(256)
void k_comp_fused(const int* __restrict__ erow, const int* __restrict__ ecol,
                  const float* __restrict__ t,
                  const unsigned short* __restrict__ dpWt,
                  const float* __restrict__ dpb,
                  float* __restrict__ tnext) {
    __shared__ __align__(16) unsigned short As[16][HH + 8];   // stride 264 (x2B, 16B mult)
    __shared__ int rows_s[16];
    int tid = threadIdx.x;
    int e0  = blockIdx.x * 16;
    if (tid < 16) rows_s[tid] = erow[e0 + tid];

    // fill: thread -> (edge r, 16 features at c0)
    int r  = tid >> 4;
    int c0 = (tid & 15) * 16;
    int re = erow[e0 + r];
    int ce = ecol[e0 + r];
    const float* pr = t + (size_t)re * HH + c0;
    const float* pc = t + (size_t)ce * HH + c0;
#pragma unroll
    for (int k = 0; k < 16; ++k) As[r][c0 + k] = f2bf(pr[k] * pc[k]);
    __syncthreads();

    int wave = tid >> 5, lane = tid & 31;
    int mb = (lane >> 4) << 3;
    int nl = lane & 15;

    // cache all 8 A K-fragments once (LDS -> VGPRs)
    v16bf afrag[8];
#pragma unroll
    for (int k = 0; k < 8; ++k)
        afrag[k] = load_frag(&As[0][0], HH + 8, 0, k * 32, lane);

#pragma unroll
    for (int half = 0; half < 2; ++half) {        // 8 waves x 2 N-tiles = 256 cols
        int n0 = (wave + half * 8) * 16;
        v8f acc = {};
#pragma unroll
        for (int k = 0; k < 8; ++k) {
            v16bf b = load_frag(dpWt, HH, n0, k * 32, lane);
            acc = wmma_bf16(afrag[k], b, acc);
        }
        float bv = dpb[n0 + nl];
#pragma unroll
        for (int rr = 0; rr < 8; ++rr) {
            int node = rows_s[mb + rr];
            atomicAdd(&tnext[(size_t)node * HH + n0 + nl], acc[rr] + bv);
        }
    }
}

// ---------------- small utility kernels ----------------
__global__ void k_cast_bf16(const float* __restrict__ s, unsigned short* __restrict__ d, int n) {
    int i = blockIdx.x * blockDim.x + threadIdx.x;
    if (i < n) d[i] = f2bf(s[i]);
}
// dst[n*K+k] = bf16(src[k*N+n])   (src is [K,N] f32, dst is [N,K] bf16)
__global__ void k_transpose_bf16(const float* __restrict__ src, unsigned short* __restrict__ dst,
                                 int K, int N) {
    int i = blockIdx.x * blockDim.x + threadIdx.x;
    if (i >= K * N) return;
    int n = i / K, k = i - n * K;
    dst[i] = f2bf(src[(size_t)k * N + n]);
}
__global__ void k_bn_stats(const float* __restrict__ y, float* __restrict__ mu,
                           float* __restrict__ rs, int M) {
    int c = blockIdx.x;
    float s1 = 0.f, s2 = 0.f;
    for (int r0 = threadIdx.x; r0 < M; r0 += blockDim.x) {
        float v = y[(size_t)r0 * HH + c];
        s1 += v; s2 += v * v;
    }
    __shared__ float sh1[256], sh2[256];
    sh1[threadIdx.x] = s1; sh2[threadIdx.x] = s2;
    __syncthreads();
    for (int s = 128; s > 0; s >>= 1) {
        if (threadIdx.x < s) { sh1[threadIdx.x] += sh1[threadIdx.x + s];
                               sh2[threadIdx.x] += sh2[threadIdx.x + s]; }
        __syncthreads();
    }
    if (threadIdx.x == 0) {
        float m = sh1[0] / M;
        float var = sh2[0] / M - m * m;
        mu[c] = m; rs[c] = rsqrtf(var + EPS);
    }
}
__global__ void k_bn_apply(const float* __restrict__ y, const float* __restrict__ mu,
                           const float* __restrict__ rs, const float* __restrict__ g,
                           const float* __restrict__ b, float* __restrict__ t,
                           unsigned short* __restrict__ tb, int n) {
    int i = blockIdx.x * blockDim.x + threadIdx.x;
    if (i >= n) return;
    int c = i & (HH - 1);
    float v = (y[i] - mu[c]) * rs[c] * g[c] + b[c];
    v = fmaxf(v, 0.0f);
    t[i] = v; tb[i] = f2bf(v);
}
__global__ void k_fill1(float* d, int n) {
    int i = blockIdx.x * blockDim.x + threadIdx.x;
    if (i < n) d[i] = 1.0f;
}
__global__ void k_deg_accum(const int* __restrict__ dst, float* __restrict__ deg, int E) {
    int i = blockIdx.x * blockDim.x + threadIdx.x;
    if (i < E) atomicAdd(&deg[dst[i]], 1.0f);
}
__global__ void k_dinv(float* d, int n) {
    int i = blockIdx.x * blockDim.x + threadIdx.x;
    if (i < n) d[i] = rsqrtf(fmaxf(d[i], 1.0f));
}
// hacc[v,f] = dinv[v]^2 * xw[v,f]   (self loop, also the init)
__global__ void k_selfloop(const float* __restrict__ xw, const float* __restrict__ dinv,
                           float* __restrict__ hacc, int n) {
    int i = blockIdx.x * blockDim.x + threadIdx.x;
    if (i >= n) return;
    int v = i >> 8;  // /HH
    float di = dinv[v];
    hacc[i] = di * di * xw[i];
}
// hacc[dst,f] += dinv[s]*dinv[d] * xw[src,f] ; 4 features per thread
__global__ void k_edge_agg(const int* __restrict__ src, const int* __restrict__ dst,
                           const float* __restrict__ dinv, const float* __restrict__ xw,
                           float* __restrict__ hacc) {
    int i = blockIdx.x * blockDim.x + threadIdx.x;   // E * 64 threads
    int e  = i >> 6;
    int f0 = (i & 63) << 2;
    if (e >= EE) return;
    int s = src[e], d = dst[e];
    float nrm = dinv[s] * dinv[d];
    const float* xs = xw + (size_t)s * HH + f0;
    float* hd = hacc + (size_t)d * HH + f0;
#pragma unroll
    for (int k = 0; k < 4; ++k) atomicAdd(&hd[k], nrm * xs[k]);
}
__global__ void k_relu_bias(float* __restrict__ h, const float* __restrict__ b, int n) {
    int i = blockIdx.x * blockDim.x + threadIdx.x;
    if (i < n) h[i] = fmaxf(h[i] + b[i & (HH - 1)], 0.0f);
}
// p[b] = [t[s], t[d], t[s]*t[d]] in bf16
__global__ void k_pair_build(const float* __restrict__ t, const int* __restrict__ sid,
                             const int* __restrict__ did, unsigned short* __restrict__ p) {
    int i = blockIdx.x * blockDim.x + threadIdx.x;   // BB*HH threads
    int b = i >> 8, f = i & (HH - 1);
    if (b >= BB) return;
    float vs = t[(size_t)sid[b] * HH + f];
    float vd = t[(size_t)did[b] * HH + f];
    unsigned short* pb = p + (size_t)b * (3 * HH);
    pb[f]          = f2bf(vs);
    pb[HH + f]     = f2bf(vd);
    pb[2 * HH + f] = f2bf(vs * vd);
}
// out[b,c] = sum_k relu(outs[b,k]) * W[k,c] + bias[c]
__global__ void k_classifier(const float* __restrict__ outs, const float* __restrict__ W,
                             const float* __restrict__ bias, float* __restrict__ out) {
    int i = blockIdx.x * blockDim.x + threadIdx.x;   // BB*CC
    if (i >= BB * CC) return;
    int b = i / CC, c = i - b * CC;
    const float* ob = outs + (size_t)b * KC;
    float acc = bias[c];
    for (int k = 0; k < KC; ++k) acc += fmaxf(ob[k], 0.0f) * W[k * CC + c];
    out[i] = acc;
}

// ---------------- host orchestration ----------------
static inline int cdiv(int a, int b) { return (a + b - 1) / b; }

extern "C" void kernel_launch(void* const* d_in, const int* in_sizes, int n_in,
                              void* d_out, int out_size, void* d_ws, size_t ws_size,
                              hipStream_t stream) {
    const float* x        = (const float*)d_in[0];
    const int*   edges    = (const int*)d_in[1];
    const int*   node_id  = (const int*)d_in[2];
    const float* lin0_W   = (const float*)d_in[3];
    const float* lin0_b   = (const float*)d_in[4];
    const float* bn_g     = (const float*)d_in[5];
    const float* bn_b     = (const float*)d_in[6];
    const float* gcn_W    = (const float*)d_in[7];
    const float* gcn_b    = (const float*)d_in[8];
    const float* dp_W     = (const float*)d_in[9];
    const float* dp_b     = (const float*)d_in[10];
    const float* fc1_W    = (const float*)d_in[11];
    const float* fc1_b    = (const float*)d_in[12];
    const float* fc2_W    = (const float*)d_in[13];
    const float* fc2_b    = (const float*)d_in[14];
    const float* fc3_W    = (const float*)d_in[15];
    const float* fc3_b    = (const float*)d_in[16];
    const float* cls_W    = (const float*)d_in[17];
    const float* cls_b    = (const float*)d_in[18];
    float* out = (float*)d_out;

    // workspace carve-out
    char* base = (char*)d_ws;
    size_t off = 0;
    auto carve = [&](size_t bytes) -> char* {
        char* p = base + off;
        off += (bytes + 255) & ~(size_t)255;
        return p;
    };
    unsigned short* xb   = (unsigned short*)carve((size_t)NN * DD * 2);
    float*          bufA = (float*)carve((size_t)NN * HH * 4);
    float*          bufB = (float*)carve((size_t)NN * HH * 4);
    unsigned short* tb   = (unsigned short*)carve((size_t)NN * HH * 2);
    float*          xw   = (float*)carve((size_t)NN * HH * 4);   // also lin0 output y
    float*          dinv = (float*)carve((size_t)NN * 4);
    float*          mu   = (float*)carve(HH * 4);
    float*          rs   = (float*)carve(HH * 4);
    unsigned short* wt   = (unsigned short*)carve((size_t)F1 * 3 * HH * 2); // max transpose (fc1)
    unsigned short* pB   = (unsigned short*)carve((size_t)BB * 3 * HH * 2);
    unsigned short* q1   = (unsigned short*)carve((size_t)BB * F1 * 2);
    unsigned short* q2   = (unsigned short*)carve((size_t)BB * F2 * 2);
    float*          outs = (float*)carve((size_t)BB * KC * 4);
    (void)ws_size; (void)in_sizes; (void)n_in; (void)out_size;

    const int T = 256;
    const int MT_NODES = cdiv(NN / 16, 8);   // 157 (20000 = 1250*16)
    const int MT_PAIRS = cdiv(BB / 16, 8);   // 64

    // x -> bf16 (once)
    k_cast_bf16<<<cdiv(NN * DD, T), T, 0, stream>>>(x, xb, NN * DD);

    for (int i = 0; i < CC; ++i) {
        const int* erow = edges + ((size_t)i * 2 + 0) * EE;   // src / row
        const int* ecol = edges + ((size_t)i * 2 + 1) * EE;   // dst / col

        // ---- lin0 + BN + ReLU ----
        k_transpose_bf16<<<cdiv(DD * HH, T), T, 0, stream>>>(lin0_W + (size_t)i * DD * HH, wt, DD, HH);
        k_gemm_bf16<4><<<dim3(HH / 64, MT_NODES), T, 0, stream>>>(
            xb, wt, xw, nullptr, lin0_b + (size_t)i * HH, NN, DD, HH, 0, 0);
        k_bn_stats<<<HH, T, 0, stream>>>(xw, mu, rs, NN);
        k_bn_apply<<<cdiv(NN * HH, T), T, 0, stream>>>(
            xw, mu, rs, bn_g + (size_t)i * HH, bn_b + (size_t)i * HH, bufA, tb, NN * HH);

        // ---- degrees (shared across both layers of this class) ----
        k_fill1<<<cdiv(NN, T), T, 0, stream>>>(dinv, NN);           // self-loop => deg starts at 1
        k_deg_accum<<<cdiv(EE, T), T, 0, stream>>>(ecol, dinv, EE);
        k_dinv<<<cdiv(NN, T), T, 0, stream>>>(dinv, NN);

        float* tcur = bufA;
        float* tnxt = bufB;
        for (int j = 0; j < LL; ++j) {
            const float* gW = gcn_W + ((size_t)i * LL + j) * HH * HH;
            const float* gB = gcn_b + ((size_t)i * LL + j) * HH;
            const float* dW = dp_W  + ((size_t)i * LL + j) * HH * HH;
            const float* dB = dp_b  + ((size_t)i * LL + j) * HH;

            // xw = t @ gW
            k_transpose_bf16<<<cdiv(HH * HH, T), T, 0, stream>>>(gW, wt, HH, HH);
            k_gemm_bf16<4><<<dim3(HH / 64, MT_NODES), T, 0, stream>>>(
                tb, wt, xw, nullptr, nullptr, NN, HH, HH, 0, 0);
            // aggregate: tnxt = D^-1/2 A D^-1/2 xw  (self-loop init + edge atomics)
            k_selfloop<<<cdiv(NN * HH, T), T, 0, stream>>>(xw, dinv, tnxt, NN * HH);
            k_edge_agg<<<cdiv(EE * 64, T), T, 0, stream>>>(erow, ecol, dinv, xw, tnxt);
            // h = relu(agg + b)
            k_relu_bias<<<cdiv(NN * HH, T), T, 0, stream>>>(tnxt, gB, NN * HH);
            // tnxt += segment_sum((t[col]*t[row]) @ dW + dB, row)   (fused WMMA)
            k_transpose_bf16<<<cdiv(HH * HH, T), T, 0, stream>>>(dW, wt, HH, HH);
            k_comp_fused<<<EE / 16, T, 0, stream>>>(erow, ecol, tcur, wt, dB, tnxt);
            // bf16 copy for next layer's GEMM
            k_cast_bf16<<<cdiv(NN * HH, T), T, 0, stream>>>(tnxt, tb, NN * HH);
            float* tmp = tcur; tcur = tnxt; tnxt = tmp;
        }

        // ---- pair head ----
        k_pair_build<<<cdiv(BB * HH, T), T, 0, stream>>>(tcur, node_id, node_id + BB, pB);
        k_transpose_bf16<<<cdiv(3 * HH * F1, T), T, 0, stream>>>(fc1_W + (size_t)i * 3 * HH * F1, wt, 3 * HH, F1);
        k_gemm_bf16<4><<<dim3(F1 / 64, MT_PAIRS), T, 0, stream>>>(
            pB, wt, nullptr, q1, fc1_b + (size_t)i * F1, BB, 3 * HH, F1, 0, 1);
        k_transpose_bf16<<<cdiv(F1 * F2, T), T, 0, stream>>>(fc2_W + (size_t)i * F1 * F2, wt, F1, F2);
        k_gemm_bf16<4><<<dim3(F2 / 64, MT_PAIRS), T, 0, stream>>>(
            q1, wt, nullptr, q2, fc2_b + (size_t)i * F2, BB, F1, F2, 0, 1);
        k_transpose_bf16<<<cdiv(F2 * F3, T), T, 0, stream>>>(fc3_W + (size_t)i * F2 * F3, wt, F2, F3);
        k_gemm_bf16<1><<<dim3(F3 / 16, MT_PAIRS), T, 0, stream>>>(
            q2, wt, outs, nullptr, fc3_b + (size_t)i * F3, BB, F2, KC, i * F3, 0);
    }

    // final classifier: relu(outs) @ cls_W + cls_b
    k_classifier<<<cdiv(BB * CC, T), T, 0, stream>>>(outs, cls_W, cls_b, out);
}
